// MultiHeadSocialAttention_88562225644178
// MI455X (gfx1250) — matrix operand order repaired and checked by reading, not compiled, not run on hardware
//
#include <hip/hip_runtime.h>
#include <hip/hip_bf16.h>

#define HID    256
#define MAXNB  64
#define NTOTAL 131072
#define BQ     4096
#define LDH    264   // LDS row stride in halfs (16B aligned, bank-rotated)

typedef _Float16 v16h __attribute__((ext_vector_type(16)));
typedef _Float16 v8h  __attribute__((ext_vector_type(8)));
typedef _Float16 v4h  __attribute__((ext_vector_type(4)));
typedef float    v8f  __attribute__((ext_vector_type(8)));

// ---- WMMA fragment helpers (16x16x32 f16; this code's contiguous k = 16*(lane/16)+s in both A and B) ----

// A-matrix 16x32 (MxK): lane holds row M=lane%16; halfs K = k0+off+{0..15},
// off = (lane>=16 ? 16 : 0). Two 16B loads.
__device__ __forceinline__ v16h load_a_frag(const _Float16* base, int m0, int k0,
                                            int ld, int lane) {
  int row  = m0 + (lane & 15);
  int koff = k0 + ((lane & 16) ? 16 : 0);
  const _Float16* p = base + row * ld + koff;
  v8h lo = *(const v8h*)(p);
  v8h hi = *(const v8h*)(p + 8);
  v16h a;
#pragma unroll
  for (int j = 0; j < 8; ++j) { a[j] = lo[j]; a[j + 8] = hi[j]; }
  return a;
}

// B-matrix 32x16 (KxN): B[k][n] = W[n][k] (W row-major [out][in]).
// lane holds col N=lane%16, halfs K = k0+{0..15} (+16 for lanes 16..31):
// one contiguous 32B load from the weight row.
__device__ __forceinline__ v16h load_b_frag(const _Float16* W, int n0, int k0,
                                            int ld, int lane) {
  int col  = n0 + (lane & 15);
  int koff = k0 + ((lane & 16) ? 16 : 0);
  return *(const v16h*)(W + col * ld + koff);
}

#define WMMA_F16(a, b, c) \
  __builtin_amdgcn_wmma_f32_16x16x32_f16(false, (a), false, (b), (short)0, (c), false, false)

// ---- f32 -> f16 converter (vectorized) ----
__global__ void cvt_f32_f16_kernel(const float* __restrict__ src,
                                   _Float16* __restrict__ dst, int n4) {
  int i = blockIdx.x * blockDim.x + threadIdx.x;
  if (i < n4) {
    float4 v = ((const float4*)src)[i];
    v4h h;
    h[0] = (_Float16)v.x; h[1] = (_Float16)v.y;
    h[2] = (_Float16)v.z; h[3] = (_Float16)v.w;
    ((v4h*)dst)[i] = h;
  }
}

// ---- generic WMMA GEMM: out[M,256] = act(A[M,256] @ W[256,256]^T + bias)*alpha
// one 16x64 tile per wave (A-frag reused across 4 WMMAs per k-step) ----
__global__ __launch_bounds__(256) void wmma_gemm_kernel(
    const _Float16* __restrict__ A, const _Float16* __restrict__ W,
    const float* __restrict__ bias, float* __restrict__ out,
    int nItems, float alpha, int do_relu) {
  int lane = threadIdx.x & 31;
  int item = blockIdx.x * 8 + (threadIdx.x >> 5);
  if (item >= nItems) return;          // wave-uniform
  int m0 = (item >> 2) << 4;           // 4 n-groups of 64 cols each
  int ng = (item & 3) << 6;
  v8f c0 = {}, c1 = {}, c2 = {}, c3 = {};
#pragma unroll
  for (int kk = 0; kk < 8; ++kk) {
    int k0 = kk * 32;
    v16h a  = load_a_frag(A, m0, k0, HID, lane);
    v16h b0 = load_b_frag(W, ng +  0, k0, HID, lane);
    v16h b1 = load_b_frag(W, ng + 16, k0, HID, lane);
    v16h b2 = load_b_frag(W, ng + 32, k0, HID, lane);
    v16h b3 = load_b_frag(W, ng + 48, k0, HID, lane);
    c0 = WMMA_F16(a, b0, c0);
    c1 = WMMA_F16(a, b1, c1);
    c2 = WMMA_F16(a, b2, c2);
    c3 = WMMA_F16(a, b3, c3);
  }
  int rbase = m0 + ((lane & 16) ? 8 : 0);
  const v8f* acc[4] = {&c0, &c1, &c2, &c3};
#pragma unroll
  for (int q = 0; q < 4; ++q) {
    int col  = ng + q * 16 + (lane & 15);
    float bb = bias[col];
#pragma unroll
    for (int r = 0; r < 8; ++r) {
      float v = (*acc[q])[r] + bb;
      if (do_relu) v = v > 0.f ? v : 0.f;
      out[(rbase + r) * HID + col] = v * alpha;
    }
  }
}

// ---- fused per-sample attention: ht gather -> K/V WMMA projections (only
// the ceil(len/16) live m-tiles; B-frags loaded once per k-step and reused
// across all m-tiles) -> masked softmax -> attn @ V.
// One workgroup (8 waves) per sample; wave w handles (mat=w>>2, cols w&3). ----
__global__ __launch_bounds__(256) void social_attn_kernel(
    const float* __restrict__ social, const int* __restrict__ starts,
    const int* __restrict__ ends, const _Float16* __restrict__ Wk_h,
    const _Float16* __restrict__ Wv_h, const float* __restrict__ bk,
    const float* __restrict__ bv, const float* __restrict__ Qbuf,
    _Float16* __restrict__ Obuf) {
  extern __shared__ __align__(16) char smem[];
  _Float16* htl = (_Float16*)smem;          // [64][LDH] f16 neighbor feats
  _Float16* Kl  = htl + MAXNB * LDH;        // [64][LDH] f16 K
  _Float16* Vl  = Kl + MAXNB * LDH;         // [64][LDH] f16 V
  float* qld = (float*)(Vl + MAXNB * LDH);  // [256] query (post relu/scale)
  float* scr = qld + HID;                   // [4][64] exp weights
  float* inv = scr + HID;                   // [4] 1/sum

  const int b    = blockIdx.x;
  const int t    = threadIdx.x;
  const int lane = t & 31;
  const int wave = t >> 5;
  const int s    = starts[b];
  int len = ends[b] - s;
  if (len < 1) len = 1;
  if (len > MAXNB) len = MAXNB;
  const int mt    = (len + 15) >> 4;        // live 16-row m-tiles (1..4)
  const int nrows = mt << 4;

  // stage only the live ht rows [nrows,256] f32 -> f16 LDS (float4 granularity)
  for (int i = t; i < nrows * (HID / 4); i += 256) {
    int row = i >> 6;            // 64 float4 per row
    int c4  = (i & 63) * 4;
    int gr  = s + row;
    if (gr > NTOTAL - 1) gr = NTOTAL - 1;
    float4 v = *(const float4*)(social + (size_t)gr * HID + c4);
    v4h h;
    h[0] = (_Float16)v.x; h[1] = (_Float16)v.y;
    h[2] = (_Float16)v.z; h[3] = (_Float16)v.w;
    *(v4h*)(htl + row * LDH + c4) = h;
  }
  qld[t] = Qbuf[(size_t)b * HID + t];
  __syncthreads();

  // K/V projections. wave -> (mat, 64-col group). Per k-step: load 4 B-frags
  // once, stream up to 4 A-frags (live m-tiles) through 4..16 WMMAs.
  {
    const int mat = wave >> 2;
    const int ng  = (wave & 3) << 6;
    const _Float16* W    = mat ? Wv_h : Wk_h;
    const float*    bias = mat ? bv : bk;
    _Float16*       dst  = mat ? Vl : Kl;
    v8f acc[4][4] = {};                      // [m-tile][n-tile]
#pragma unroll
    for (int kk = 0; kk < 8; ++kk) {
      int k0 = kk * 32;
      v16h b0 = load_b_frag(W, ng +  0, k0, HID, lane);
      v16h b1 = load_b_frag(W, ng + 16, k0, HID, lane);
      v16h b2 = load_b_frag(W, ng + 32, k0, HID, lane);
      v16h b3 = load_b_frag(W, ng + 48, k0, HID, lane);
#pragma unroll
      for (int mtl = 0; mtl < 4; ++mtl) {
        if (mtl < mt) {                      // workgroup-uniform: EXEC stays ~0
          v16h a = load_a_frag(htl, mtl << 4, k0, LDH, lane);
          acc[mtl][0] = WMMA_F16(a, b0, acc[mtl][0]);
          acc[mtl][1] = WMMA_F16(a, b1, acc[mtl][1]);
          acc[mtl][2] = WMMA_F16(a, b2, acc[mtl][2]);
          acc[mtl][3] = WMMA_F16(a, b3, acc[mtl][3]);
        }
      }
    }
#pragma unroll
    for (int mtl = 0; mtl < 4; ++mtl) {
      if (mtl < mt) {
        int rbase = (mtl << 4) + ((lane & 16) ? 8 : 0);
#pragma unroll
        for (int q = 0; q < 4; ++q) {
          int col  = ng + q * 16 + (lane & 15);
          float bb = bias[col];
#pragma unroll
          for (int r = 0; r < 8; ++r) {
            float v = acc[mtl][q][r] + bb;
            v = v > 0.f ? v : 0.f;           // relu
            dst[(rbase + r) * LDH + col] = (_Float16)v;
          }
        }
      }
    }
  }
  __syncthreads();

  // scores[h][m] = q_h . K[m]_h for live m only
  {
    int h = t >> 6, m = t & 63;
    float acc = -1e30f;
    if (m < len) {
      const _Float16* kp = Kl + m * LDH + h * 64;
      const float*    qp = qld + h * 64;
      acc = 0.f;
#pragma unroll 8
      for (int d = 0; d < 64; ++d) acc += qp[d] * (float)kp[d];
    }
    scr[t] = acc;
  }
  __syncthreads();

  // per-head softmax normalizers over live m (4 heads)
  if (t < 4) {
    float mx = -3.0e38f;
    for (int m = 0; m < len; ++m) mx = fmaxf(mx, scr[t * 64 + m]);
    float sum = 0.f;
    for (int m = 0; m < len; ++m) {
      float e = __expf(scr[t * 64 + m] - mx);
      scr[t * 64 + m] = e;
      sum += e;
    }
    inv[t] = 1.f / sum;
  }
  __syncthreads();

  // out[h*64+d] = (sum_{m<len} w[h][m] * V[m][h*64+d]) / sum
  {
    int h = t >> 6, d = t & 63;
    const _Float16* vp = Vl + h * 64 + d;
    const float*    ap = scr + h * 64;
    float acc = 0.f;
    for (int m = 0; m < len; ++m) acc += ap[m] * (float)vp[m * LDH];
    Obuf[(size_t)b * HID + t] = (_Float16)(acc * inv[h]);
  }
}

extern "C" void kernel_launch(void* const* d_in, const int* in_sizes, int n_in,
                              void* d_out, int out_size, void* d_ws, size_t ws_size,
                              hipStream_t stream) {
  const float* enc    = (const float*)d_in[0];
  const float* social = (const float*)d_in[1];
  const float* Wq = (const float*)d_in[2];
  const float* bq = (const float*)d_in[3];
  const float* Wk = (const float*)d_in[4];
  const float* bk = (const float*)d_in[5];
  const float* Wv = (const float*)d_in[6];
  const float* bv = (const float*)d_in[7];
  const float* Wf = (const float*)d_in[8];
  const float* bf = (const float*)d_in[9];
  const int* starts = (const int*)d_in[10];
  const int* ends   = (const int*)d_in[11];
  float* out = (float*)d_out;

  char* ws = (char*)d_ws;
  _Float16* Wq_h  = (_Float16*)(ws + 0);
  _Float16* Wk_h  = (_Float16*)(ws + (1u << 17));
  _Float16* Wv_h  = (_Float16*)(ws + (2u << 17));
  _Float16* Wf_h  = (_Float16*)(ws + (3u << 17));
  _Float16* enc_h = (_Float16*)(ws + (4u << 17));              // 2 MB
  float*    Qbuf  = (float*)   (ws + (4u << 17) + (1u << 21)); // 4 MB
  _Float16* Obuf  = (_Float16*)(ws + (4u << 17) + (1u << 21) + (1u << 22));

  // weight / activation f16 conversion
  cvt_f32_f16_kernel<<<64, 256, 0, stream>>>(Wq, Wq_h, 16384);
  cvt_f32_f16_kernel<<<64, 256, 0, stream>>>(Wk, Wk_h, 16384);
  cvt_f32_f16_kernel<<<64, 256, 0, stream>>>(Wv, Wv_h, 16384);
  cvt_f32_f16_kernel<<<64, 256, 0, stream>>>(Wf, Wf_h, 16384);
  cvt_f32_f16_kernel<<<1024, 256, 0, stream>>>(enc, enc_h, 262144);

  // Q = relu(enc @ Wq^T + bq) / sqrt(256): 1024 16x64 items -> 128 blocks
  wmma_gemm_kernel<<<128, 256, 0, stream>>>(enc_h, Wq_h, bq, Qbuf,
                                            (BQ / 16) * 4, 0.0625f, 1);

  // fused gather + K/V projection + attention
  size_t smemBytes = (size_t)(3 * MAXNB * LDH) * sizeof(_Float16) +
                     (size_t)(HID + HID + 8) * sizeof(float);
  (void)hipFuncSetAttribute((const void*)social_attn_kernel,
                            hipFuncAttributeMaxDynamicSharedMemorySize,
                            (int)smemBytes);
  social_attn_kernel<<<BQ, 256, smemBytes, stream>>>(
      social, starts, ends, Wk_h, Wv_h, bk, bv, Qbuf, Obuf);

  // final projection: out = O @ Wf^T + bf
  wmma_gemm_kernel<<<128, 256, 0, stream>>>(Obuf, Wf_h, bf, out,
                                            (BQ / 16) * 4, 1.0f, 0);
}
